// DenseToSparseWrapper_24464133718818
// MI455X (gfx1250) — compile-verified, matrix operand order checked by reading, not run on hardware
//
#include <hip/hip_runtime.h>
#include <hip/hip_bf16.h>
#include <stdint.h>

// ---------------------------------------------------------------------------
// Problem constants (from reference): B=32, N=128, D=32, H=64, C=10, P=2H+1
// ---------------------------------------------------------------------------
#define BB 32
#define NN 128
#define DD 32
#define HH 64
#define CCLS 10
#define TPB 256

typedef __bf16 bf16_t;
typedef __attribute__((ext_vector_type(16))) __bf16 v16bf;
typedef __attribute__((ext_vector_type(8)))  __bf16 v8bf;
typedef __attribute__((ext_vector_type(4)))  __bf16 v4bf;
typedef __attribute__((ext_vector_type(8)))  float  v8f;

// ---------------------------------------------------------------------------
// WMMA helpers (CDNA5 wave32, V_WMMA_F32_16X16X32_BF16)
// A-frag (ISA 7.12.2): lane L = row (L&15); element e -> K = (e&7) +
// ((e>>3)<<4) + ((L>=16)?8:0): two contiguous 16B runs -> 2x ds_load_b128.
// B operands staged transposed in LDS so the same loader serves both.
// ---------------------------------------------------------------------------
__device__ __forceinline__ v16bf load_frag(const bf16_t* p, int ld, int lane) {
    int r  = lane & 15;
    int kb = (lane >> 4) << 3;
    const bf16_t* q = p + r * ld + kb;
    union { uint4 u[2]; v16bf v; } u;
    u.u[0] = *(const uint4*)(q);
    u.u[1] = *(const uint4*)(q + 16);
    return u.v;
}

__device__ __forceinline__ v8f wmma_bf16(v16bf a, v16bf b, v8f c) {
    return __builtin_amdgcn_wmma_f32_16x16x32_bf16(false, a, false, b,
                                                   (short)0, c, false, false);
}

__device__ __forceinline__ v8f zero8() {
    v8f z;
#pragma unroll
    for (int e = 0; e < 8; ++e) z[e] = 0.f;
    return z;
}

// ---------------------------------------------------------------------------
// Async global->LDS copy (GLOBAL_LOAD_ASYNC_TO_LDS_B128, ASYNCcnt tracked).
// Low 32 bits of a generic LDS pointer are the wave-relative LDS byte offset.
// ---------------------------------------------------------------------------
__device__ __forceinline__ void async_copy16(void* ldsp, const void* gptr) {
    uint32_t off = (uint32_t)(uintptr_t)ldsp;
    asm volatile("global_load_async_to_lds_b128 %0, %1, off"
                 :: "v"(off), "v"(gptr) : "memory");
}
__device__ __forceinline__ void wait_async() {
    asm volatile("s_wait_asynccnt 0x0" ::: "memory");
}

__device__ __forceinline__ void f4arr(float4 v, float* o) {
    o[0] = v.x; o[1] = v.y; o[2] = v.z; o[3] = v.w;
}

// ---------------------------------------------------------------------------
// One-time conversion of adj to bf16: adjF keeps the diagonal (GCN layers),
// adjC zeroes it (pair layers). Fully vectorized: b128 in, b64 out x2.
// ---------------------------------------------------------------------------
__global__ void k_prep(const float* __restrict__ adj,
                       bf16_t* __restrict__ adjF, bf16_t* __restrict__ adjC) {
    int idx = blockIdx.x * blockDim.x + threadIdx.x;
    if (idx >= BB * NN * NN / 4) return;
    float4 v = ((const float4*)adj)[idx];
    float vals[4]; f4arr(v, vals);
    int ij = (idx * 4) % (NN * NN);
    int i = ij / NN, j0 = ij % NN;
    v4bf f, c;
#pragma unroll
    for (int q = 0; q < 4; ++q) {
        f[q] = (bf16_t)vals[q];
        c[q] = (i == j0 + q) ? (bf16_t)0.f : f[q];
    }
    ((v4bf*)adjF)[idx] = f;
    ((v4bf*)adjC)[idx] = c;
}

// ---------------------------------------------------------------------------
// Generic small linear: Y[r,o] = sum_d X[r,d]*W[d,o] (+ bias[o])
// ---------------------------------------------------------------------------
__global__ void k_lin(const float* __restrict__ X, const float* __restrict__ W,
                      const float* __restrict__ bias, float* __restrict__ Y,
                      int R, int Din, int Dout) {
    int idx = blockIdx.x * blockDim.x + threadIdx.x;
    if (idx >= R * Dout) return;
    int r = idx / Dout, o = idx % Dout;
    float acc = bias ? bias[o] : 0.f;
    const float* xr = X + (size_t)r * Din;
#pragma unroll 8
    for (int d = 0; d < Din; ++d) acc += xr[d] * W[d * Dout + o];
    Y[idx] = acc;
}

// ---------------------------------------------------------------------------
// GCN combine: h = relu(tSelf + adj @ tNbr), one batch per workgroup.
// adj slabs staged via async b128 copies from pre-converted bf16.
// ---------------------------------------------------------------------------
__global__ void k_gcn(const bf16_t* __restrict__ adjF,
                      const float* __restrict__ tSelf,
                      const float* __restrict__ tNbr,
                      float* __restrict__ hOut) {
    __shared__ __align__(16) bf16_t Gt[HH * NN];   // B^T: [d][w]  16 KB
    __shared__ __align__(16) bf16_t As[NN * 32];   // A slab       8 KB
    int b = blockIdx.x, t = threadIdx.x;
    const bf16_t* adjb = adjF + (size_t)b * NN * NN;

    for (int e = t; e < NN * HH / 4; e += TPB) {   // stage tNbr transposed
        int d = e >> 5, w0 = (e & 31) * 4;
        v4bf g;
#pragma unroll
        for (int q = 0; q < 4; ++q)
            g[q] = (bf16_t)tNbr[((size_t)b * NN + w0 + q) * HH + d];
        *(v4bf*)(&Gt[d * NN + w0]) = g;
    }
    int wave = t >> 5, lane = t & 31, r0 = wave * 16;
    v8f c[4];
#pragma unroll
    for (int q = 0; q < 4; ++q) c[q] = zero8();

    for (int kk = 0; kk < NN; kk += 32) {
        __syncthreads();
#pragma unroll
        for (int c2 = 0; c2 < 2; ++c2) {           // 512 x 16B chunks
            int chunk = t + c2 * TPB;
            int i = chunk >> 2, cc = chunk & 3;
            async_copy16(&As[i * 32 + cc * 8], adjb + i * NN + kk + cc * 8);
        }
        wait_async();
        __syncthreads();
        v16bf a = load_frag(&As[r0 * 32], 32, lane);
#pragma unroll
        for (int nt = 0; nt < 4; ++nt) {
            v16bf bf = load_frag(&Gt[(nt * 16) * NN + kk], NN, lane);
            c[nt] = wmma_bf16(a, bf, c[nt]);
        }
    }
    int n = lane & 15, mo = (lane >> 4) << 3;
#pragma unroll
    for (int nt = 0; nt < 4; ++nt)
#pragma unroll
        for (int e = 0; e < 8; ++e) {
            int m = r0 + mo + e, col = nt * 16 + n;
            size_t o = ((size_t)b * NN + m) * HH + col;
            hOut[o] = fmaxf(tSelf[o] + c[nt][e], 0.f);
        }
}

// ---------------------------------------------------------------------------
// M1[b,k] = adj_clean[b] @ G_k ; G_k[w,d] = U[min,d] + V[max,d] + a_kw*wlast[d]
// ---------------------------------------------------------------------------
__global__ void k_pairM(const bf16_t* __restrict__ adjC,
                        const float* __restrict__ U, const float* __restrict__ V,
                        const float* __restrict__ wlast,
                        bf16_t* __restrict__ M) {
    __shared__ __align__(16) bf16_t Gt[HH * NN];
    __shared__ __align__(16) bf16_t As[NN * 32];
    int k = blockIdx.x, b = blockIdx.y, t = threadIdx.x;
    const bf16_t* adjb = adjC + (size_t)b * NN * NN;
    const float* Ub = U + (size_t)b * NN * HH;
    const float* Vb = V + (size_t)b * NN * HH;

    for (int e = t; e < NN * HH / 4; e += TPB) {   // build Gt, 4-wide in w
        int d = e >> 5, w0 = (e & 31) * 4;
        v4bf av = *(const v4bf*)(adjb + k * NN + w0);   // diag pre-zeroed
        float wl = wlast[d];
        v4bf g;
#pragma unroll
        for (int q = 0; q < 4; ++q) {
            int w = w0 + q, mn = min(k, w), mx = max(k, w);
            g[q] = (bf16_t)(Ub[mn * HH + d] + Vb[mx * HH + d] + (float)av[q] * wl);
        }
        *(v4bf*)(&Gt[d * NN + w0]) = g;
    }
    int wave = t >> 5, lane = t & 31, r0 = wave * 16;
    v8f c[4];
#pragma unroll
    for (int q = 0; q < 4; ++q) c[q] = zero8();

    for (int kk = 0; kk < NN; kk += 32) {
        __syncthreads();
#pragma unroll
        for (int c2 = 0; c2 < 2; ++c2) {
            int chunk = t + c2 * TPB;
            int i = chunk >> 2, cc = chunk & 3;
            async_copy16(&As[i * 32 + cc * 8], adjb + i * NN + kk + cc * 8);
        }
        wait_async();
        __syncthreads();
        v16bf a = load_frag(&As[r0 * 32], 32, lane);
#pragma unroll
        for (int nt = 0; nt < 4; ++nt) {
            v16bf bf = load_frag(&Gt[(nt * 16) * NN + kk], NN, lane);
            c[nt] = wmma_bf16(a, bf, c[nt]);
        }
    }
    bf16_t* Mk = M + (((size_t)b * NN) + k) * (size_t)(NN * HH);
    int n = lane & 15, mo = (lane >> 4) << 3;
#pragma unroll
    for (int nt = 0; nt < 4; ++nt)
#pragma unroll
        for (int e = 0; e < 8; ++e)
            Mk[(r0 + mo + e) * HH + nt * 16 + n] = (bf16_t)c[nt][e];
}

// ---------------------------------------------------------------------------
// M2[b,k] = adj_clean[b] @ ( relu(pair1_k) @ p2w2 + b2 ), chained WMMAs.
// pair1_k rebuilt 8-wide in d from M1 (b128 loads) + low-rank self terms.
// ---------------------------------------------------------------------------
__global__ void k_M2(const bf16_t* __restrict__ adjC,
                     const float* __restrict__ U1, const float* __restrict__ V1,
                     const float* __restrict__ w1last,
                     const bf16_t* __restrict__ M1,
                     const float* __restrict__ p2w2, const float* __restrict__ p2b2,
                     bf16_t* __restrict__ M2) {
    __shared__ __align__(16) bf16_t P1[NN * HH];    // [w][d]  16 KB
    __shared__ __align__(16) bf16_t W2t[HH * HH];   // [o][d]   8 KB
    __shared__ __align__(16) bf16_t Gt[HH * NN];    // [o][w]  16 KB
    __shared__ __align__(16) bf16_t As[NN * 32];    //          8 KB
    int k = blockIdx.x, b = blockIdx.y, t = threadIdx.x;
    const bf16_t* adjb = adjC + (size_t)b * NN * NN;
    const float* U1b = U1 + (size_t)b * NN * HH;
    const float* V1b = V1 + (size_t)b * NN * HH;
    const bf16_t* M1b = M1 + (size_t)b * NN * (size_t)(NN * HH);

    for (int e = t; e < NN * HH / 8; e += TPB) {    // rebuild pair1 row-block
        int w = e >> 3, d0 = (e & 7) * 8;
        int mn = min(k, w), mx = max(k, w);
        float a = (float)adjb[k * NN + w];
        float uu[8], vv[8], wl[8];
        f4arr(*(const float4*)(U1b + mn * HH + d0),     uu);
        f4arr(*(const float4*)(U1b + mn * HH + d0 + 4), uu + 4);
        f4arr(*(const float4*)(V1b + mx * HH + d0),     vv);
        f4arr(*(const float4*)(V1b + mx * HH + d0 + 4), vv + 4);
        f4arr(*(const float4*)(w1last + d0),            wl);
        f4arr(*(const float4*)(w1last + d0 + 4),        wl + 4);
        v8bf m1a = *(const v8bf*)(M1b + ((size_t)w * NN + k) * HH + d0);
        v8bf m1c = *(const v8bf*)(M1b + ((size_t)k * NN + w) * HH + d0);
        v8bf outp;
#pragma unroll
        for (int q = 0; q < 8; ++q) {
            float s = uu[q] + vv[q] + a * wl[q] + (float)m1a[q] + (float)m1c[q];
            outp[q] = (bf16_t)fmaxf(s, 0.f);
        }
        *(v8bf*)(&P1[w * HH + d0]) = outp;
    }
    for (int e = t; e < HH * HH / 4; e += TPB) {    // stage p2w2 transposed
        int o = e >> 4, d0 = (e & 15) * 4;
        v4bf wv;
#pragma unroll
        for (int q = 0; q < 4; ++q) wv[q] = (bf16_t)p2w2[(d0 + q) * HH + o];
        *(v4bf*)(&W2t[o * HH + d0]) = wv;
    }
    __syncthreads();

    int wave = t >> 5, lane = t & 31, r0 = wave * 16;
    int n = lane & 15, mo = (lane >> 4) << 3;

    // G2 = P1 @ p2w2 + b2
    v8f c[4];
#pragma unroll
    for (int q = 0; q < 4; ++q) c[q] = zero8();
#pragma unroll
    for (int kk = 0; kk < HH; kk += 32) {
        v16bf a = load_frag(&P1[r0 * HH + kk], HH, lane);
#pragma unroll
        for (int nt = 0; nt < 4; ++nt) {
            v16bf bf = load_frag(&W2t[(nt * 16) * HH + kk], HH, lane);
            c[nt] = wmma_bf16(a, bf, c[nt]);
        }
    }
#pragma unroll
    for (int nt = 0; nt < 4; ++nt) {                // transposed b128 store
        int o = nt * 16 + n;
        float bias = p2b2[o];
        v8bf g;
#pragma unroll
        for (int e = 0; e < 8; ++e) g[e] = (bf16_t)(c[nt][e] + bias);
        *(v8bf*)(&Gt[o * NN + r0 + mo]) = g;
    }

    // M2_k = adj_clean @ G2
    v8f acc[4];
#pragma unroll
    for (int q = 0; q < 4; ++q) acc[q] = zero8();
    for (int kk = 0; kk < NN; kk += 32) {
        __syncthreads();
#pragma unroll
        for (int c2 = 0; c2 < 2; ++c2) {
            int chunk = t + c2 * TPB;
            int i = chunk >> 2, cc = chunk & 3;
            async_copy16(&As[i * 32 + cc * 8], adjb + i * NN + kk + cc * 8);
        }
        wait_async();
        __syncthreads();
        v16bf a = load_frag(&As[r0 * 32], 32, lane);
#pragma unroll
        for (int nt = 0; nt < 4; ++nt) {
            v16bf bf = load_frag(&Gt[(nt * 16) * NN + kk], NN, lane);
            acc[nt] = wmma_bf16(a, bf, acc[nt]);
        }
    }
    bf16_t* Mk = M2 + (((size_t)b * NN) + k) * (size_t)(NN * HH);
#pragma unroll
    for (int nt = 0; nt < 4; ++nt)
#pragma unroll
        for (int e = 0; e < 8; ++e)
            Mk[(r0 + mo + e) * HH + nt * 16 + n] = (bf16_t)acc[nt][e];
}

// ---------------------------------------------------------------------------
// emb2[b] += sum_{j>i} relu( pair1_i @ p2w1 + S2 ),
// S2[j][d] = M2[b,j][i,d] + M2[b,i][j,d] + p2b1[d]  (precomputed, b128 loads)
// ---------------------------------------------------------------------------
__global__ void k_emb2(const bf16_t* __restrict__ adjC,
                       const float* __restrict__ U1, const float* __restrict__ V1,
                       const float* __restrict__ w1last,
                       const bf16_t* __restrict__ M1, const bf16_t* __restrict__ M2,
                       const float* __restrict__ p2w1, const float* __restrict__ p2b1,
                       float* __restrict__ emb2) {
    __shared__ __align__(16) bf16_t P1[NN * HH];    // 16 KB
    __shared__ __align__(16) bf16_t W1t[HH * HH];   //  8 KB
    __shared__ __align__(16) float  S2[NN * HH];    // 32 KB
    __shared__ float redbuf[HH];
    int i = blockIdx.x, b = blockIdx.y, t = threadIdx.x;
    const bf16_t* adjb = adjC + (size_t)b * NN * NN;
    const float* U1b = U1 + (size_t)b * NN * HH;
    const float* V1b = V1 + (size_t)b * NN * HH;
    const bf16_t* M1b = M1 + (size_t)b * NN * (size_t)(NN * HH);
    const bf16_t* M2b = M2 + (size_t)b * NN * (size_t)(NN * HH);

    if (t < HH) redbuf[t] = 0.f;
    for (int e = t; e < NN * HH / 8; e += TPB) {
        int w = e >> 3, d0 = (e & 7) * 8;
        int mn = min(i, w), mx = max(i, w);
        float a = (float)adjb[i * NN + w];
        float uu[8], vv[8], wl[8], bb[8];
        f4arr(*(const float4*)(U1b + mn * HH + d0),     uu);
        f4arr(*(const float4*)(U1b + mn * HH + d0 + 4), uu + 4);
        f4arr(*(const float4*)(V1b + mx * HH + d0),     vv);
        f4arr(*(const float4*)(V1b + mx * HH + d0 + 4), vv + 4);
        f4arr(*(const float4*)(w1last + d0),            wl);
        f4arr(*(const float4*)(w1last + d0 + 4),        wl + 4);
        f4arr(*(const float4*)(p2b1 + d0),              bb);
        f4arr(*(const float4*)(p2b1 + d0 + 4),          bb + 4);
        v8bf m1a = *(const v8bf*)(M1b + ((size_t)w * NN + i) * HH + d0);
        v8bf m1c = *(const v8bf*)(M1b + ((size_t)i * NN + w) * HH + d0);
        v8bf m2a = *(const v8bf*)(M2b + ((size_t)w * NN + i) * HH + d0);
        v8bf m2c = *(const v8bf*)(M2b + ((size_t)i * NN + w) * HH + d0);
        v8bf outp;
        float s2[8];
#pragma unroll
        for (int q = 0; q < 8; ++q) {
            float s = uu[q] + vv[q] + a * wl[q] + (float)m1a[q] + (float)m1c[q];
            outp[q] = (bf16_t)fmaxf(s, 0.f);
            s2[q] = (float)m2a[q] + (float)m2c[q] + bb[q];
        }
        *(v8bf*)(&P1[w * HH + d0]) = outp;
        *(float4*)(&S2[w * HH + d0])     = make_float4(s2[0], s2[1], s2[2], s2[3]);
        *(float4*)(&S2[w * HH + d0 + 4]) = make_float4(s2[4], s2[5], s2[6], s2[7]);
    }
    for (int e = t; e < HH * HH / 4; e += TPB) {
        int o = e >> 4, d0 = (e & 15) * 4;
        v4bf wv;
#pragma unroll
        for (int q = 0; q < 4; ++q) wv[q] = (bf16_t)p2w1[(d0 + q) * HH + o];
        *(v4bf*)(&W1t[o * HH + d0]) = wv;
    }
    __syncthreads();

    int wave = t >> 5, lane = t & 31, r0 = wave * 16;
    int n = lane & 15, mo = (lane >> 4) << 3;

    v8f c[4];
#pragma unroll
    for (int q = 0; q < 4; ++q) c[q] = zero8();
#pragma unroll
    for (int kk = 0; kk < HH; kk += 32) {
        v16bf a = load_frag(&P1[r0 * HH + kk], HH, lane);
#pragma unroll
        for (int nt = 0; nt < 4; ++nt) {
            v16bf bf = load_frag(&W1t[(nt * 16) * HH + kk], HH, lane);
            c[nt] = wmma_bf16(a, bf, c[nt]);
        }
    }
#pragma unroll
    for (int nt = 0; nt < 4; ++nt) {
        int d = nt * 16 + n;
        float s = 0.f;
#pragma unroll
        for (int e = 0; e < 8; ++e) {
            int j = r0 + mo + e;
            float v = fmaxf(c[nt][e] + S2[j * HH + d], 0.f);
            if (j > i) s += v;               // strict upper triangle
        }
        atomicAdd(&redbuf[d], s);            // ds_add_f32
    }
    __syncthreads();
    if (t < HH) atomicAdd(&emb2[b * HH + t], redbuf[t]);
}

// ---------------------------------------------------------------------------
// Small tail kernels
// ---------------------------------------------------------------------------
__global__ void k_zero(float* p, int n) {
    int i = blockIdx.x * blockDim.x + threadIdx.x;
    if (i < n) p[i] = 0.f;
}

__global__ void k_emb1(const float* __restrict__ h, float* __restrict__ emb1) {
    int idx = blockIdx.x * blockDim.x + threadIdx.x;
    if (idx >= BB * HH) return;
    int b = idx / HH, d = idx % HH;
    float s = 0.f;
    for (int i = 0; i < NN; ++i) s += h[((size_t)b * NN + i) * HH + d];
    emb1[idx] = s;
}

__global__ void k_final(const float* __restrict__ emb1, const float* __restrict__ emb2,
                        const float* __restrict__ cw, const float* __restrict__ cb,
                        float* __restrict__ out) {
    int idx = blockIdx.x * blockDim.x + threadIdx.x;
    if (idx >= BB * CCLS) return;
    int b = idx / CCLS, c = idx % CCLS;
    float acc = cb[c];
#pragma unroll 8
    for (int d = 0; d < HH; ++d) acc += emb1[b * HH + d] * cw[d * CCLS + c];
#pragma unroll 8
    for (int d = 0; d < HH; ++d) acc += emb2[b * HH + d] * cw[(HH + d) * CCLS + c];
    out[idx] = acc;
}

// ---------------------------------------------------------------------------
// Host-side orchestration
// ---------------------------------------------------------------------------
extern "C" void kernel_launch(void* const* d_in, const int* in_sizes, int n_in,
                              void* d_out, int out_size, void* d_ws, size_t ws_size,
                              hipStream_t stream) {
    const float* x    = (const float*)d_in[0];
    const float* adj  = (const float*)d_in[1];
    const float* w11  = (const float*)d_in[2];
    const float* b11  = (const float*)d_in[3];
    const float* w12  = (const float*)d_in[4];
    const float* b12  = (const float*)d_in[5];
    const float* w21  = (const float*)d_in[6];
    const float* b21  = (const float*)d_in[7];
    const float* w22  = (const float*)d_in[8];
    const float* b22  = (const float*)d_in[9];
    const float* p1w1 = (const float*)d_in[10];
    const float* p1b1 = (const float*)d_in[11];
    const float* p1w2 = (const float*)d_in[12];
    const float* p1b2 = (const float*)d_in[13];
    const float* p2w1 = (const float*)d_in[14];
    const float* p2b1 = (const float*)d_in[15];
    const float* p2w2 = (const float*)d_in[16];
    const float* p2b2 = (const float*)d_in[17];
    const float* cw   = (const float*)d_in[18];
    const float* cb   = (const float*)d_in[19];
    float* out = (float*)d_out;

    char* ws = (char*)d_ws;
    const size_t SZ = (size_t)BB * NN * HH * sizeof(float);   // 1 MiB
    float* tA   = (float*)(ws + 0 * SZ);
    float* tB   = (float*)(ws + 1 * SZ);
    float* hA   = (float*)(ws + 2 * SZ);
    float* hB   = (float*)(ws + 3 * SZ);
    float* U1   = (float*)(ws + 4 * SZ);
    float* V1   = (float*)(ws + 5 * SZ);
    float* U2   = (float*)(ws + 6 * SZ);
    float* V2   = (float*)(ws + 7 * SZ);
    float* emb1 = (float*)(ws + 8 * SZ);
    float* emb2 = (float*)(ws + 8 * SZ + 32768);
    bf16_t* adjF = (bf16_t*)(ws + 9 * SZ);                    // 1 MiB
    bf16_t* adjC = (bf16_t*)(ws + 10 * SZ);                   // 1 MiB
    const size_t MSZ = (size_t)BB * NN * NN * HH * sizeof(bf16_t); // 64 MiB
    bf16_t* M1 = (bf16_t*)(ws + 11 * SZ);
    bf16_t* M2 = (bf16_t*)(ws + 11 * SZ + MSZ);

    const int RNH = BB * NN * HH;                    // 262144
    dim3 gLin((RNH + TPB - 1) / TPB), blk(TPB);
    dim3 gBatch(BB);
    dim3 gPair(NN, BB);

    // adj -> bf16 (with diag / diag-zeroed), once per call
    k_prep<<<dim3((BB * NN * NN / 4 + TPB - 1) / TPB), blk, 0, stream>>>(adj, adjF, adjC);

    // GCN layer 1
    k_lin<<<gLin, blk, 0, stream>>>(x, w11, b11, tA, BB * NN, DD, HH);
    k_lin<<<gLin, blk, 0, stream>>>(x, w12, b12, tB, BB * NN, DD, HH);
    k_gcn<<<gBatch, blk, 0, stream>>>(adjF, tA, tB, hA);
    // GCN layer 2
    k_lin<<<gLin, blk, 0, stream>>>(hA, w21, b21, tA, BB * NN, HH, HH);
    k_lin<<<gLin, blk, 0, stream>>>(hA, w22, b22, tB, BB * NN, HH, HH);
    k_gcn<<<gBatch, blk, 0, stream>>>(adjF, tA, tB, hB);

    // Low-rank decomposition of pair-layer-1 weights applied to h
    k_lin<<<gLin, blk, 0, stream>>>(hB, p1w1,           p1b1,    U1, BB * NN, HH, HH);
    k_lin<<<gLin, blk, 0, stream>>>(hB, p1w1 + HH * HH, nullptr, V1, BB * NN, HH, HH);
    k_lin<<<gLin, blk, 0, stream>>>(hB, p1w2,           p1b2,    U2, BB * NN, HH, HH);
    k_lin<<<gLin, blk, 0, stream>>>(hB, p1w2 + HH * HH, nullptr, V2, BB * NN, HH, HH);

    // Aggregation matmul families (the heavy WMMA work)
    k_pairM<<<gPair, blk, 0, stream>>>(adjC, U2, V2, p1w2 + 2 * HH * HH, M1);
    k_M2<<<gPair, blk, 0, stream>>>(adjC, U1, V1, p1w1 + 2 * HH * HH, M1,
                                    p2w2, p2b2, M2);

    // emb2 reduction (zero accumulator every call: harness poisons ws)
    k_zero<<<dim3((BB * HH + TPB - 1) / TPB), blk, 0, stream>>>(emb2, BB * HH);
    k_emb2<<<gPair, blk, 0, stream>>>(adjC, U1, V1, p1w1 + 2 * HH * HH, M1, M2,
                                      p2w1, p2b1, emb2);

    // emb1 and classifier
    k_emb1<<<dim3((BB * HH + TPB - 1) / TPB), blk, 0, stream>>>(hB, emb1);
    k_final<<<dim3((BB * CCLS + TPB - 1) / TPB), blk, 0, stream>>>(emb1, emb2, cw, cb, out);
}
